// KNRM_2147483648565
// MI455X (gfx1250) — compile-verified, matrix operand order checked by reading, not run on hardware
//
#include <hip/hip_runtime.h>

typedef __attribute__((ext_vector_type(16))) _Float16 v16h;
typedef __attribute__((ext_vector_type(8)))  _Float16 v8h;
typedef __attribute__((ext_vector_type(8)))  float    v8f;

#define LQ     32
#define LD     512
#define EDIM   50
#define EHALF  25            // EDIM/2 float2 loads per row
#define KPAD   64
#define NKER   21
#define DCHUNK 128

#define LOG2E_10   14.4269504089f      // 10 * log2(e)
#define LOG2E_50   72.1347520444f      // 50 * log2(e)
#define LOG2E_EX   721347.520444f      // 500000 * log2(e)

// Raw v_exp_f32 (exp2), no denormal guard: far-tail underflow flushes to 0,
// which is exactly the behavior we want for gaussian tails.
#define EXP2RAW(x) __builtin_amdgcn_exp2f(x)

// e^{-1} .. e^{-9}: shared constant ladder of both gaussian recurrence chains
__device__ __constant__ float EXP_NEG[10] = {
    1.0f,                 0.36787944117f,  0.13533528324f,  0.04978706836f,
    0.01831563889f,       0.00673794700f,  0.00247875218f,  0.00091188197f,
    0.00033546263f,       0.00012340980f };

// Build a 16-half WMMA fragment from LDS: per-lane halves are K[kh..kh+7] and
// K[kh+16..kh+23] of one matrix row (16-bit A/B VGPR layout, wave32).
__device__ __forceinline__ v16h ld_frag(const _Float16* p) {
    v8h lo = *(const v8h*)(p);
    v8h hi = *(const v8h*)(p + 16);
    return __builtin_shufflevector(lo, hi,
        0, 1, 2, 3, 4, 5, 6, 7, 8, 9, 10, 11, 12, 13, 14, 15);
}

// Gather one embedding row (float2 vector loads), L2-normalize, store f16 K-padded.
__device__ __forceinline__ void gather_row(const float* __restrict__ emb, int row,
                                           _Float16* __restrict__ dst) {
    const float2* src2 = (const float2*)(emb + row * EDIM);
    float2 v[EHALF];
    float ss = 0.0f;
    #pragma unroll
    for (int i = 0; i < EHALF; ++i) {
        v[i] = src2[i];
        ss += v[i].x * v[i].x + v[i].y * v[i].y;
    }
    const float rn = (ss > 0.0f) ? rsqrtf(ss) : 0.0f;   // NaN guard: 0-row -> sim 0
    #pragma unroll
    for (int i = 0; i < EHALF; ++i) {
        dst[2 * i]     = (_Float16)(v[i].x * rn);
        dst[2 * i + 1] = (_Float16)(v[i].y * rn);
    }
    #pragma unroll
    for (int i = EDIM; i < KPAD; ++i) dst[i] = (_Float16)0.0f;
}

__global__ __launch_bounds__(256) void knrm_fused(
    const int* __restrict__ q1, const int* __restrict__ d1,
    const int* __restrict__ q2, const int* __restrict__ d2,
    const float* __restrict__ emb,
    const float* __restrict__ mlp_w, const float* __restrict__ mlp_b,
    float* __restrict__ out)
{
    __shared__ _Float16 Qh[LQ * KPAD];        //  4 KB  normalized query tile
    __shared__ _Float16 Dh[DCHUNK * KPAD];    // 16 KB  normalized doc chunk
    __shared__ float    sumG[NKER * LQ];      // per-(kernel, query) gaussian sums
    __shared__ float    logitAcc;

    const int b     = blockIdx.x;
    const int tid   = threadIdx.x;
    const int lane  = tid & 31;
    const int wave  = tid >> 5;
    const int l16   = lane & 15;
    const int khalf = (lane < 16) ? 0 : 8;    // K-half split of the wave32 frag layout

    float logit[2];

    for (int p = 0; p < 2; ++p) {
        const int* qidx = p ? q2 : q1;
        const int* didx = p ? d2 : d1;

        for (int i = tid; i < NKER * LQ; i += 256) sumG[i] = 0.0f;
        if (tid == 0) logitAcc = 0.0f;

        // ---- gather + L2-normalize query rows into LDS (f16, K padded to 64)
        if (tid < LQ)
            gather_row(emb, qidx[b * LQ + tid], &Qh[tid * KPAD]);
        __syncthreads();

        // ---- B operand (queries, N dimension): fixed per wave, kept in regs
        const int qt   = wave & 1;           // query tile 0/1
        const int qrow = qt * 16 + l16;
        const v16h bfrag0 = ld_frag(&Qh[qrow * KPAD +  0 + khalf]);
        const v16h bfrag1 = ld_frag(&Qh[qrow * KPAD + 32 + khalf]);

        // per-lane accumulators: sum over docs of gaussian k, for this lane's query
        float acc[NKER];
        #pragma unroll
        for (int k = 0; k < NKER; ++k) acc[k] = 0.0f;

        for (int cc = 0; cc < LD / DCHUNK; ++cc) {
            __syncthreads();                 // previous chunk fully consumed
            if (tid < DCHUNK)
                gather_row(emb, didx[b * LD + cc * DCHUNK + tid], &Dh[tid * KPAD]);
            __syncthreads();

            const int g = wave >> 1;         // 4 wave-groups per query tile
            #pragma unroll
            for (int dt0 = 0; dt0 < 2; ++dt0) {
                const int dt   = g * 2 + dt0;        // doc tile within chunk
                const int drow = dt * 16 + l16;

                v8f c = {};
                const v16h a0 = ld_frag(&Dh[drow * KPAD +  0 + khalf]);
                c = __builtin_amdgcn_wmma_f32_16x16x32_f16(
                        false, a0, false, bfrag0, (short)0, c, false, false);
                const v16h a1 = ld_frag(&Dh[drow * KPAD + 32 + khalf]);
                c = __builtin_amdgcn_wmma_f32_16x16x32_f16(
                        false, a1, false, bfrag1, (short)0, c, false, false);

                // c[r] = cosine(doc dt*16 + r(+8), query qt*16 + l16).
                // Equal-sigma gaussian ladder: 4 raw v_exp_f32 instead of 21 exps.
                //   t    = e^{10s},  tinv = e^{-10s}  (exp2 with source-neg)
                //   g_10 = e^{-50(s-0.05)^2}          (seed, in [e^-56, 1])
                //   g_9  = g_10 * tinv                (exact identity)
                //   g_{10+j} = g_{10+j-1} * t    * e^{-j}
                //   g_{9-j}  = g_{9-j+1}  * tinv * e^{-j}
                #pragma unroll
                for (int r = 0; r < 8; ++r) {
                    const float s    = c[r];
                    const float ts   = LOG2E_10 * s;
                    const float t    = EXP2RAW( ts);
                    const float tinv = EXP2RAW(-ts);

                    const float du = s - 0.05f;
                    float gu = EXP2RAW(-LOG2E_50 * du * du); // g_10
                    float gd = gu * tinv;                    // g_9
                    acc[10] += gu;
                    acc[9]  += gd;
                    #pragma unroll
                    for (int j = 1; j <= 9; ++j) {
                        gu *= t    * EXP_NEG[j];             // g_{10+j}
                        gd *= tinv * EXP_NEG[j];             // g_{9-j}
                        acc[10 + j] += gu;
                        acc[9 - j]  += gd;
                    }
                    // exact-match kernel (sigma = 0.001)
                    const float de = s - 1.0f;
                    acc[20] += EXP2RAW(-LOG2E_EX * de * de);
                }
            }
        }

        // lanes L and L^16 hold the same query column: fold, then LDS-atomic merge
        #pragma unroll
        for (int k = 0; k < NKER; ++k) {
            const float v = acc[k] + __shfl_xor(acc[k], 16, 32);
            if (lane < 16) atomicAdd(&sumG[k * LQ + qt * 16 + l16], v);
        }
        __syncthreads();

        // pooling: logit = sum_{k,q} w[k] * log1p(sumG[k][q]) + bias
        float part = 0.0f;
        for (int i = tid; i < NKER * LQ; i += 256)
            part += mlp_w[i >> 5] * __logf(1.0f + sumG[i]);
        #pragma unroll
        for (int off = 16; off > 0; off >>= 1)
            part += __shfl_xor(part, off, 32);
        if (lane == 0) atomicAdd(&logitAcc, part);
        __syncthreads();
        logit[p] = logitAcc + mlp_b[0];
        __syncthreads();                     // LDS reused by next pair
    }

    if (tid == 0) {
        const float x = logit[0] - logit[1];
        out[b] = 1.0f / (1.0f + __expf(-x));
    }
}

extern "C" void kernel_launch(void* const* d_in, const int* in_sizes, int n_in,
                              void* d_out, int out_size, void* d_ws, size_t ws_size,
                              hipStream_t stream) {
    (void)n_in; (void)out_size; (void)d_ws; (void)ws_size;
    const int*   q1  = (const int*)d_in[0];
    const int*   d1  = (const int*)d_in[1];
    const int*   q2  = (const int*)d_in[2];
    const int*   d2  = (const int*)d_in[3];
    const float* emb = (const float*)d_in[4];
    const float* w   = (const float*)d_in[5];
    const float* bb  = (const float*)d_in[6];
    float*       out = (float*)d_out;

    const int B = in_sizes[0] / LQ;          // 1024
    knrm_fused<<<dim3(B), dim3(256), 0, stream>>>(q1, d1, q2, d2, emb, w, bb, out);
}
